// FFN_GCNs_13572096655679
// MI455X (gfx1250) — matrix operand order repaired and checked by reading, not compiled, not run on hardware
//
#include <hip/hip_runtime.h>
#include <hip/hip_bf16.h>

// MI455X / gfx1250, wave32. See analysis: NNConv restructured so the heavy
// per-edge transform becomes a per-node GEMM done with V_WMMA_F32_16X16X32_F16,
// with Mnode kept in f16 (41MB, L2-resident) for the edge gather phase.

typedef __attribute__((ext_vector_type(16))) _Float16 v16h;
typedef __attribute__((ext_vector_type(8)))  float    v8f;

#define IN_DIM  32
#define HID     32
#define NNODES  20000
#define NEDGES  200000
#define NPAIRS  100000
#define NODE_BLOCKS (NNODES / 16)   // 1250, exact

// ---------------------------------------------------------------------------
// Prep: permute w2 [32][1024] (w2[k][i*32+o]) into WMMA B-operand tile order.
// GEMM computed:  Mnode_flat[n][c] = sum_i x[n][i] * Bsrc[i][c],  c = o*32+k.
// B tile t, lane l, element m holds Bsrc[K(l,m)][t*16+(l&15)] with
// K(l,m) = m + 16*(l>=16)  (contiguous k order chosen here, matching the A load).
// Also convert b2 into two B tiles for the per-node bias vector bn.
// ---------------------------------------------------------------------------
__global__ __launch_bounds__(256)
void k_prep(const float* __restrict__ w2, const float* __restrict__ b2,
            _Float16* __restrict__ Bp,   // [64 tiles][32 lanes][16] f16
            _Float16* __restrict__ B2p)  // [ 2 tiles][32 lanes][16] f16
{
    int u = blockIdx.x * blockDim.x + threadIdx.x;
    if (u < 64 * 32 * 16) {
        int t = u >> 9, l = (u >> 4) & 31, m = u & 15;
        int i = m + 16 * (l >> 4);        // GEMM K index == x feature index
        int c = t * 16 + (l & 15);        // flattened output column = o*32+k
        int o = c >> 5, k = c & 31;
        Bp[u] = (_Float16)w2[k * (IN_DIM * HID) + i * HID + o];
    }
    int u2 = u - 64 * 32 * 16;
    if (u2 >= 0 && u2 < 2 * 32 * 16) {
        int t = u2 >> 9, l = (u2 >> 4) & 31, m = u2 & 15;
        int i = m + 16 * (l >> 4);
        int o = t * 16 + (l & 15);
        B2p[u2] = (_Float16)b2[i * HID + o];
    }
}

// ---------------------------------------------------------------------------
// Node transform GEMM: Mnode[n][o][k] = sum_i x[n][i]*w2[k][i*32+o]  (f16 out)
//                      bn[n][o]       = sum_i x[n][i]*b2[i*32+o]     (f32 out)
// One wave per 16-node block; A loaded once, 64+2 WMMA tiles per wave.
// A operand as loaded here (contiguous k, same order as Bp): lane row = lane&15,
//   elements 0..7  = x[row][16*hi ..],  elements 8..15 = x[row][16*hi+8 ..]
// D layout: lane col = lane&15, vgpr v -> row v + 8*hi.
// ---------------------------------------------------------------------------
__global__ __launch_bounds__(256)
void k_node_mats(const float* __restrict__ x,
                 const _Float16* __restrict__ Bp,
                 const _Float16* __restrict__ B2p,
                 _Float16* __restrict__ Mnode,   // [N][1024], o-major k-minor
                 float* __restrict__ bn)         // [N][32]
{
    int wave = threadIdx.x >> 5;
    int lane = threadIdx.x & 31;
    int nodeBlock = blockIdx.x * 8 + wave;
    if (nodeBlock >= NODE_BLOCKS) return;       // wave-uniform; EXEC stays full
    int hi  = lane >> 4;
    int row = nodeBlock * 16 + (lane & 15);
    const float* xr = x + row * IN_DIM;

    v16h a;
#pragma unroll
    for (int m = 0; m < 8; ++m) a[m]     = (_Float16)xr[16 * hi + m];
#pragma unroll
    for (int m = 0; m < 8; ++m) a[8 + m] = (_Float16)xr[16 * hi + 8 + m];

    int rbase = nodeBlock * 16 + 8 * hi;

    for (int t = 0; t < 64; ++t) {
        v16h b = *(const v16h*)(Bp + ((t * 32 + lane) << 4));  // 32B/lane, linear
        v8f  c = {};
        c = __builtin_amdgcn_wmma_f32_16x16x32_f16(false, a, false, b,
                                                   (short)0, c, false, false);
        int col = t * 16 + (lane & 15);
        _Float16* mp = Mnode + (size_t)rbase * 1024 + col;
#pragma unroll
        for (int v = 0; v < 8; ++v) mp[(size_t)v * 1024] = (_Float16)c[v];
    }
    for (int t = 0; t < 2; ++t) {
        v16h b = *(const v16h*)(B2p + ((t * 32 + lane) << 4));
        v8f  c = {};
        c = __builtin_amdgcn_wmma_f32_16x16x32_f16(false, a, false, b,
                                                   (short)0, c, false, false);
        int col = t * 16 + (lane & 15);
        float* bp = bn + (size_t)rbase * 32 + col;
#pragma unroll
        for (int v = 0; v < 8; ++v) bp[(size_t)v * 32] = c[v];
    }
}

// ---------------------------------------------------------------------------
// Edge phase: one wave per edge. lane o computes
//   msg[o] = bn[src][o] + sum_k h[k] * Mnode[src][o][k]
// h[k] lives in lane k (edge MLP), broadcast via wave32 shuffles.
// Mnode row slice: each lane reads 32 contiguous f16 (64B); the wave streams
// a contiguous 2KB L2-resident row. Scatter-add via global f32 atomics.
// ---------------------------------------------------------------------------
__global__ __launch_bounds__(256)
void k_edge(const int* __restrict__ ei, const float* __restrict__ ea,
            const float* __restrict__ w1, const float* __restrict__ b1,
            const _Float16* __restrict__ Mnode, const float* __restrict__ bn,
            float* __restrict__ agg)
{
    int e = blockIdx.x * 8 + (threadIdx.x >> 5);
    if (e >= NEDGES) return;                     // wave-uniform
    int lane = threadIdx.x & 31;

    float ea0 = ea[2 * e], ea1 = ea[2 * e + 1];
    float h = ea0 * w1[lane] + ea1 * w1[32 + lane] + b1[lane];
    h = h > 0.f ? h : 0.f;

    int src = ei[e];
    int dst = ei[NEDGES + e];

    const v16h* mp = (const v16h*)(Mnode + (size_t)src * 1024 + lane * 32);
    v16h m0 = mp[0], m1 = mp[1];

    float msg = bn[src * 32 + lane];
#pragma unroll
    for (int k = 0; k < 16; ++k) msg += __shfl(h, k,      32) * (float)m0[k];
#pragma unroll
    for (int k = 0; k < 16; ++k) msg += __shfl(h, 16 + k, 32) * (float)m1[k];

    atomicAdd(agg + (size_t)dst * 32 + lane, msg);
}

// out[n][o] = relu(agg[n][o] + sum_i xin[n][i]*root[i][o] + bias[o])
__global__ __launch_bounds__(256)
void k_finish(const float* __restrict__ agg, const float* __restrict__ xin,
              const float* __restrict__ root, const float* __restrict__ bias,
              float* __restrict__ xout)
{
    int t = blockIdx.x * blockDim.x + threadIdx.x;
    if (t >= NNODES * 32) return;
    int n = t >> 5, o = t & 31;
    float acc = agg[t] + bias[o];
    const float* xr = xin + n * 32;
#pragma unroll
    for (int i = 0; i < 32; ++i) acc += xr[i] * root[i * 32 + o];
    xout[t] = acc > 0.f ? acc : 0.f;
}

__global__ __launch_bounds__(256)
void k_zero(float* __restrict__ p, int n)
{
    int t = blockIdx.x * blockDim.x + threadIdx.x;
    if (t < n) p[t] = 0.f;
}

// FC head: out[p] = relu(concat(xl[label0], xr[label1]) @ fcw + fcb)
__global__ __launch_bounds__(256)
void k_fc(const int* __restrict__ label, const float* __restrict__ xl,
          const float* __restrict__ xr, const float* __restrict__ fcw,
          const float* __restrict__ fcb, float* __restrict__ out)
{
    int p = blockIdx.x * blockDim.x + threadIdx.x;
    if (p >= NPAIRS) return;
    int ia = label[2 * p], ib = label[2 * p + 1];
    const float* za = xl + (size_t)ia * 32;
    const float* zb = xr + (size_t)ib * 32;
    float o0 = fcb[0], o1 = fcb[1];
#pragma unroll
    for (int d = 0; d < 32; ++d) { float v = za[d]; o0 += v * fcw[2*d];        o1 += v * fcw[2*d+1]; }
#pragma unroll
    for (int d = 0; d < 32; ++d) { float v = zb[d]; o0 += v * fcw[2*(32+d)];   o1 += v * fcw[2*(32+d)+1]; }
    out[2 * p]     = o0 > 0.f ? o0 : 0.f;
    out[2 * p + 1] = o1 > 0.f ? o1 : 0.f;
}

extern "C" void kernel_launch(void* const* d_in, const int* in_sizes, int n_in,
                              void* d_out, int out_size, void* d_ws, size_t ws_size,
                              hipStream_t stream)
{
    (void)in_sizes; (void)n_in; (void)out_size; (void)ws_size;
    const float* x1      = (const float*)d_in[0];
    const int*   ei1     = (const int*)  d_in[1];
    const float* ea1     = (const float*)d_in[2];
    const float* x2      = (const float*)d_in[3];
    const int*   ei2     = (const int*)  d_in[4];
    const float* ea2     = (const float*)d_in[5];
    const int*   label   = (const int*)  d_in[6];
    const float* l1_w1   = (const float*)d_in[7];
    const float* l1_b1   = (const float*)d_in[8];
    const float* l1_w2   = (const float*)d_in[9];
    const float* l1_b2   = (const float*)d_in[10];
    const float* l1_root = (const float*)d_in[11];
    const float* l1_bias = (const float*)d_in[12];
    const float* l2_w1   = (const float*)d_in[13];
    const float* l2_b1   = (const float*)d_in[14];
    const float* l2_w2   = (const float*)d_in[15];
    const float* l2_b2   = (const float*)d_in[16];
    const float* l2_root = (const float*)d_in[17];
    const float* l2_bias = (const float*)d_in[18];
    const float* fc_w    = (const float*)d_in[19];
    const float* fc_b    = (const float*)d_in[20];

    // workspace carving (aligned to 256B)
    char* w = (char*)d_ws;
    auto take = [&](size_t bytes) -> char* {
        char* p = w; w += (bytes + 255) & ~(size_t)255; return p;
    };
    _Float16* Bp1   = (_Float16*)take(64 * 32 * 16 * sizeof(_Float16)); // 64KB
    _Float16* B2p1  = (_Float16*)take( 2 * 32 * 16 * sizeof(_Float16));
    _Float16* Bp2   = (_Float16*)take(64 * 32 * 16 * sizeof(_Float16));
    _Float16* B2p2  = (_Float16*)take( 2 * 32 * 16 * sizeof(_Float16));
    _Float16* Mnode = (_Float16*)take((size_t)NNODES * 1024 * sizeof(_Float16)); // 41MB
    float*    bn    = (float*)   take((size_t)NNODES * 32 * sizeof(float));
    float*    agg   = (float*)   take((size_t)NNODES * 32 * sizeof(float));
    float*    xtmp  = (float*)   take((size_t)NNODES * 32 * sizeof(float));
    float*    xf1   = (float*)   take((size_t)NNODES * 32 * sizeof(float));
    float*    xf2   = (float*)   take((size_t)NNODES * 32 * sizeof(float));

    const int prepThreads = 64 * 32 * 16 + 2 * 32 * 16;     // 33792
    k_prep<<<(prepThreads + 255) / 256, 256, 0, stream>>>(l1_w2, l1_b2, Bp1, B2p1);
    k_prep<<<(prepThreads + 255) / 256, 256, 0, stream>>>(l2_w2, l2_b2, Bp2, B2p2);

    const int matBlocks  = (NODE_BLOCKS + 7) / 8;           // 157
    const int zeroBlocks = (NNODES * 32 + 255) / 256;       // 2500
    const int edgeBlocks = NEDGES / 8;                      // 25000

    for (int g = 0; g < 2; ++g) {
        const float* xin = g ? x2  : x1;
        const int*   ei  = g ? ei2 : ei1;
        const float* ea  = g ? ea2 : ea1;
        float*       xf  = g ? xf2 : xf1;
        // layer 1
        k_node_mats<<<matBlocks, 256, 0, stream>>>(xin, Bp1, B2p1, Mnode, bn);
        k_zero     <<<zeroBlocks, 256, 0, stream>>>(agg, NNODES * 32);
        k_edge     <<<edgeBlocks, 256, 0, stream>>>(ei, ea, l1_w1, l1_b1, Mnode, bn, agg);
        k_finish   <<<zeroBlocks, 256, 0, stream>>>(agg, xin, l1_root, l1_bias, xtmp);
        // layer 2
        k_node_mats<<<matBlocks, 256, 0, stream>>>(xtmp, Bp2, B2p2, Mnode, bn);
        k_zero     <<<zeroBlocks, 256, 0, stream>>>(agg, NNODES * 32);
        k_edge     <<<edgeBlocks, 256, 0, stream>>>(ei, ea, l2_w1, l2_b1, Mnode, bn, agg);
        k_finish   <<<zeroBlocks, 256, 0, stream>>>(agg, xtmp, l2_root, l2_bias, xf);
    }

    k_fc<<<(NPAIRS + 255) / 256, 256, 0, stream>>>(label, xf1, xf2, fc_w, fc_b,
                                                   (float*)d_out);
}
